// PointCloudTokenizerV1_28346784153894
// MI455X (gfx1250) — compile-verified, matrix operand order checked by reading, not run on hardware
//
#include <hip/hip_runtime.h>
#include <hip/hip_bf16.h>

// ---------------------------------------------------------------------------
// PointCloudTokenizer for MI455X (gfx1250, wave32, WMMA + TDM)
// ---------------------------------------------------------------------------

typedef __attribute__((ext_vector_type(16))) __bf16          v16bf;
typedef __attribute__((ext_vector_type(16))) unsigned short  v16us;
typedef __attribute__((ext_vector_type(8)))  unsigned short  v8us;
typedef __attribute__((ext_vector_type(8)))  float           v8f;
typedef __attribute__((ext_vector_type(4)))  float           v4f;
typedef __attribute__((ext_vector_type(4)))  unsigned int    v4u;
typedef __attribute__((ext_vector_type(8)))  int             v8i;
typedef __attribute__((ext_vector_type(4)))  int             v4i;

#define B_BATCH 8
#define NPTS    16384
#define FDIM    32
#define MTOK    128
#define KNN     16
#define DTOK    768

#if defined(__has_builtin)
#  if __has_builtin(__builtin_amdgcn_tensor_load_to_lds) && __has_builtin(__builtin_amdgcn_s_wait_tensorcnt)
#    define HAVE_TDM 1
#  endif
#endif
#ifndef HAVE_TDM
#  define HAVE_TDM 0
#endif

__device__ __forceinline__ unsigned short f32_to_bf16(float f) {
  unsigned int u = __builtin_bit_cast(unsigned int, f);
  u += 0x7FFFu + ((u >> 16) & 1u);          // round-to-nearest-even
  return (unsigned short)(u >> 16);
}
__device__ __forceinline__ unsigned int pack_bf16x2(float lo, float hi) {
  return (unsigned int)f32_to_bf16(lo) | ((unsigned int)f32_to_bf16(hi) << 16);
}
__device__ __forceinline__ float bf16_to_f32(unsigned short h) {
  unsigned int u = ((unsigned int)h) << 16;
  return __builtin_bit_cast(float, u);
}
__device__ __forceinline__ void st_val(float* p, float v)          { *p = v; }
__device__ __forceinline__ void st_val(unsigned short* p, float v) { *p = f32_to_bf16(v); }

// ---------------------------------------------------------------------------
// TDM: async DMA of a [rows x 32] bf16 tile (row stride = K elements) into LDS.
// D# built per CDNA5 ISA 08_async_tensor §8 (2D tile, data_size=2B, type=2).
// ---------------------------------------------------------------------------
#if HAVE_TDM
__device__ __forceinline__ void tdm_load_tile_bf16(
    unsigned short* ldsDst, const unsigned short* gsrc,
    unsigned int rowElems /*K*/, unsigned int rows, unsigned int cols)
{
  unsigned long long gaddr = (unsigned long long)(size_t)gsrc;
  unsigned int       laddr = (unsigned int)(size_t)ldsDst;   // addr[31:0] = LDS offset

  v4u g0;
  g0[0] = 1u;                                                // count=1, user descriptor
  g0[1] = laddr;                                             // lds_addr
  g0[2] = (unsigned int)(gaddr & 0xffffffffu);               // global_addr[31:0]
  g0[3] = (unsigned int)((gaddr >> 32) & 0x01ffffffu)        // global_addr[56:32]
        | (2u << 30);                                        // type = 2 ("image")

  unsigned long long td0 = cols;        // tensor_dim0 (elements in a tile row)
  unsigned long long td1 = rows;        // tensor_dim1
  unsigned long long s0  = rowElems;    // tensor_dim0_stride (elements)
  v8i g1;
  g1[0] = (int)(1u << 16);                                        // data_size=1 -> 2 bytes
  g1[1] = (int)((td0 & 0xffffu) << 16);                           // tensor_dim0[15:0]
  g1[2] = (int)(((td0 >> 16) & 0xffffu) | ((td1 & 0xffffu) << 16));
  g1[3] = (int)(((td1 >> 16) & 0xffffu) | ((cols & 0xffffu) << 16)); // tile_dim0
  g1[4] = (int)(rows & 0xffffu);                                  // tile_dim1 (tile_dim2=0)
  g1[5] = (int)(s0 & 0xffffffffu);
  g1[6] = (int)((s0 >> 32) & 0xffffu);                            // dim1_stride unused (2D)
  g1[7] = 0;

  v4i z4 = {0, 0, 0, 0};
#if defined(__clang_major__) && (__clang_major__ >= 23)
  v8i z8 = {0, 0, 0, 0, 0, 0, 0, 0};
  __builtin_amdgcn_tensor_load_to_lds(g0, g1, z4, z4, z8, 0);
#else
  __builtin_amdgcn_tensor_load_to_lds(g0, g1, z4, z4, 0);
#endif
}
#endif

// ---------------------------------------------------------------------------
// A-tile staging (256 rows x 32 k, bf16 in LDS).  Returns true if via TDM.
// ---------------------------------------------------------------------------
__device__ __forceinline__ bool stage_A(unsigned short* lA, const float* A,
                                        int K, int m0, int kk, int tid)
{
  // one full 32-element row per thread: 8x b128 global loads, 4x b128 LDS stores
  const float* src = &A[(size_t)(m0 + tid) * K + kk];
  v4u o[4];
  #pragma unroll
  for (int g = 0; g < 4; g++) {
    v4f f0 = *(const v4f*)(src + g * 8 + 0);
    v4f f1 = *(const v4f*)(src + g * 8 + 4);
    o[g][0] = pack_bf16x2(f0[0], f0[1]);
    o[g][1] = pack_bf16x2(f0[2], f0[3]);
    o[g][2] = pack_bf16x2(f1[0], f1[1]);
    o[g][3] = pack_bf16x2(f1[2], f1[3]);
  }
  v4u* dst = (v4u*)&lA[tid * 32];
  #pragma unroll
  for (int g = 0; g < 4; g++) dst[g] = o[g];
  return false;
}

__device__ __forceinline__ bool stage_A(unsigned short* lA, const unsigned short* A,
                                        int K, int m0, int kk, int tid)
{
#if HAVE_TDM
  if (tid < 32)   // wave 0 issues the DMA (waves with EXEC==0 branch around)
    tdm_load_tile_bf16(lA, &A[(size_t)m0 * K + kk], (unsigned)K, 256u, 32u);
  return true;
#else
  const v8us* src = (const v8us*)&A[(size_t)(m0 + tid) * K + kk];
  v8us* dst = (v8us*)&lA[tid * 32];
  #pragma unroll
  for (int g = 0; g < 4; g++) dst[g] = src[g];
  return false;
#endif
}

// ---------------------------------------------------------------------------
// Tiled WMMA GEMM:  C[M,N] = act(A[M,K] * W[K,N] + bias)
//   block tile 256(M) x 64(N), 256 threads = 8 waves, wave tile 64x32
//   -> 8x v_wmma_f32_16x16x32_bf16 per 32-wide K step
// ---------------------------------------------------------------------------
template <typename TIn, typename TOut, bool RELU>
__global__ __launch_bounds__(256) void gemm_bias_act(
    const TIn* __restrict__ A, const float* __restrict__ W,
    const float* __restrict__ bias, TOut* __restrict__ C,
    int M, int N, int K)
{
  __shared__ __align__(16) unsigned short lA[256 * 32];   // [m][k] bf16
  __shared__ __align__(16) unsigned short lBT[64 * 32];   // [n][k] bf16 (W transposed)

  const int tid  = threadIdx.x;
  const int lane = tid & 31;
  const int wave = tid >> 5;
  const int wm   = wave >> 1;              // 0..3  -> 64-row band
  const int wn   = wave & 1;               // 0..1  -> 32-col band
  const int m0   = blockIdx.y * 256;
  const int n0   = blockIdx.x * 64;
  const int lrow = lane & 15;
  const int koff = (lane & 16) ? 8 : 0;    // per-ISA 16-bit A/B lane layout

  v8f acc[4][2];
  #pragma unroll
  for (int mt = 0; mt < 4; mt++)
    #pragma unroll
    for (int nt = 0; nt < 2; nt++)
      #pragma unroll
      for (int r = 0; r < 8; r++) acc[mt][nt][r] = 0.0f;

  for (int kk = 0; kk < K; kk += 32) {
    __syncthreads();

    const bool viaTdm = stage_A(lA, A, K, m0, kk, tid);

    // W tile transposed: thread -> (n = tid&63, kg = tid>>6), 8 k's per thread
    {
      const int n  = tid & 63;
      const int kg = tid >> 6;
      const float* src = &W[(size_t)(kk + kg * 8) * N + n0 + n];
      v4u o;
      float w0 = src[0];
      float w1 = src[(size_t)1 * N];
      float w2 = src[(size_t)2 * N];
      float w3 = src[(size_t)3 * N];
      float w4 = src[(size_t)4 * N];
      float w5 = src[(size_t)5 * N];
      float w6 = src[(size_t)6 * N];
      float w7 = src[(size_t)7 * N];
      o[0] = pack_bf16x2(w0, w1);
      o[1] = pack_bf16x2(w2, w3);
      o[2] = pack_bf16x2(w4, w5);
      o[3] = pack_bf16x2(w6, w7);
      *(v4u*)&lBT[n * 32 + kg * 8] = o;
    }

    if (kk + 32 < K)  // hint next K stage of A into cache (global_prefetch_b8)
      __builtin_prefetch(&A[(size_t)(m0 + tid) * K + kk + 32], 0, 0);

#if HAVE_TDM
    if (viaTdm && tid < 32) __builtin_amdgcn_s_wait_tensorcnt(0);
#else
    (void)viaTdm;
#endif
    __syncthreads();

    v16bf a[4], b[2];
    #pragma unroll
    for (int mt = 0; mt < 4; mt++) {
      const unsigned short* base = &lA[(wm * 64 + mt * 16 + lrow) * 32];
      v8us lo = *(const v8us*)(base + koff);        // K = koff .. koff+7
      v8us hi = *(const v8us*)(base + 16 + koff);   // K = 16+koff ..
      v16us t;
      #pragma unroll
      for (int q = 0; q < 8; q++) { t[q] = lo[q]; t[8 + q] = hi[q]; }
      a[mt] = __builtin_bit_cast(v16bf, t);
    }
    #pragma unroll
    for (int nt = 0; nt < 2; nt++) {
      const unsigned short* base = &lBT[(wn * 32 + nt * 16 + lrow) * 32];
      v8us lo = *(const v8us*)(base + koff);
      v8us hi = *(const v8us*)(base + 16 + koff);
      v16us t;
      #pragma unroll
      for (int q = 0; q < 8; q++) { t[q] = lo[q]; t[8 + q] = hi[q]; }
      b[nt] = __builtin_bit_cast(v16bf, t);
    }
    #pragma unroll
    for (int mt = 0; mt < 4; mt++)
      #pragma unroll
      for (int nt = 0; nt < 2; nt++)
        acc[mt][nt] = __builtin_amdgcn_wmma_f32_16x16x32_bf16(
            false, a[mt], false, b[nt], (short)0, acc[mt][nt], false, false);
  }

  // epilogue: C/D layout -> VGPR r holds M=r (lanes 0-15) / M=8+r (lanes 16-31)
  const int rofs = (lane & 16) ? 8 : 0;
  #pragma unroll
  for (int mt = 0; mt < 4; mt++)
    #pragma unroll
    for (int nt = 0; nt < 2; nt++) {
      int gcol = n0 + wn * 32 + nt * 16 + lrow;
      float bv = bias[gcol];
      #pragma unroll
      for (int r = 0; r < 8; r++) {
        int grow = m0 + wm * 64 + mt * 16 + rofs + r;
        float v = acc[mt][nt][r] + bv;
        if (RELU) v = fmaxf(v, 0.0f);
        st_val(&C[(size_t)grow * N + gcol], v);
      }
    }
}

// ---------------------------------------------------------------------------
// Exact FPS, one workgroup per batch.  pts = [x,y,z, feat[:,F-1]]
// ---------------------------------------------------------------------------
__global__ __launch_bounds__(1024) void fps_kernel(
    const float* __restrict__ coords, const float* __restrict__ features,
    int* __restrict__ idx_out, float* __restrict__ cent_out)
{
  const int b   = blockIdx.x;
  const int tid = threadIdx.x;
  __shared__ float sval[1024];
  __shared__ int   sidx[1024];
  __shared__ int   sel[MTOK];
  __shared__ float lp[4];

  const size_t base = (size_t)b * NPTS;
  float mind[16], px[16], py[16], pz[16], pt[16];
  #pragma unroll
  for (int j = 0; j < 16; j++) {
    int p = j * 1024 + tid;
    px[j] = coords[(base + p) * 3 + 0];
    py[j] = coords[(base + p) * 3 + 1];
    pz[j] = coords[(base + p) * 3 + 2];
    pt[j] = features[(base + p) * FDIM + (FDIM - 1)];
    mind[j] = 3.4e38f;
  }
  if (tid == 0) sel[0] = 0;
  int last = 0;
  for (int it = 1; it < MTOK; it++) {
    if (tid == 0) {
      lp[0] = coords[(base + last) * 3 + 0];
      lp[1] = coords[(base + last) * 3 + 1];
      lp[2] = coords[(base + last) * 3 + 2];
      lp[3] = features[(base + last) * FDIM + (FDIM - 1)];
    }
    __syncthreads();
    float cx = lp[0], cy = lp[1], cz = lp[2], ct = lp[3];
    float bv = -1.0f; int bi = 0x7fffffff;
    #pragma unroll
    for (int j = 0; j < 16; j++) {
      float dx = px[j]-cx, dy = py[j]-cy, dz = pz[j]-cz, dt = pt[j]-ct;
      float d  = dx*dx + dy*dy + dz*dz + dt*dt;
      mind[j]  = fminf(mind[j], d);
      int p = j * 1024 + tid;
      if (mind[j] > bv || (mind[j] == bv && p < bi)) { bv = mind[j]; bi = p; }
    }
    sval[tid] = bv; sidx[tid] = bi;
    __syncthreads();
    for (int s = 512; s > 0; s >>= 1) {
      if (tid < s) {
        float v2 = sval[tid+s]; int i2 = sidx[tid+s];
        if (v2 > sval[tid] || (v2 == sval[tid] && i2 < sidx[tid])) { sval[tid] = v2; sidx[tid] = i2; }
      }
      __syncthreads();
    }
    last = sidx[0];
    if (tid == 0) sel[it] = last;
    __syncthreads();
  }
  if (tid < MTOK) {
    int p = sel[tid];
    idx_out[b * MTOK + tid] = p;
    cent_out[(b * MTOK + tid) * 4 + 0] = coords[(base + p) * 3 + 0];
    cent_out[(b * MTOK + tid) * 4 + 1] = coords[(base + p) * 3 + 1];
    cent_out[(b * MTOK + tid) * 4 + 2] = coords[(base + p) * 3 + 2];
    cent_out[(b * MTOK + tid) * 4 + 3] = features[(base + p) * FDIM + (FDIM - 1)];
  }
}

// ---------------------------------------------------------------------------
// KNN (K=16 smallest d2), one workgroup per (batch, centroid).
// Per-thread register top-16, then 16 argmin rounds over the LDS pool.
// ---------------------------------------------------------------------------
__global__ __launch_bounds__(256) void knn_kernel(
    const float* __restrict__ coords, const float* __restrict__ features,
    const float* __restrict__ cent, int* __restrict__ knn_out)
{
  const int bm  = blockIdx.x;          // b*MTOK + m
  const int b   = bm >> 7;
  const int tid = threadIdx.x;
  __shared__ float cv[256 * KNN];
  __shared__ int   ci[256 * KNN];
  __shared__ float rv[256];
  __shared__ int   ri[256];
  __shared__ int   rp[256];

  const size_t base = (size_t)b * NPTS;
  const float cx = cent[bm*4+0], cy = cent[bm*4+1], cz = cent[bm*4+2], ct = cent[bm*4+3];

  float tv[KNN]; int ti[KNN];
  #pragma unroll
  for (int k = 0; k < KNN; k++) { tv[k] = 3.4e38f; ti[k] = 0x7fffffff; }

  for (int i = 0; i < NPTS / 256; i++) {
    int p = i * 256 + tid;
    float dx = coords[(base+p)*3+0] - cx;
    float dy = coords[(base+p)*3+1] - cy;
    float dz = coords[(base+p)*3+2] - cz;
    float dt = features[(base+p)*FDIM + (FDIM-1)] - ct;
    float d  = dx*dx + dy*dy + dz*dz + dt*dt;
    if (d < tv[KNN-1]) {                       // insertion keeps ties stable
      int pos = KNN - 1;
      while (pos > 0 && tv[pos-1] > d) { tv[pos] = tv[pos-1]; ti[pos] = ti[pos-1]; pos--; }
      tv[pos] = d; ti[pos] = p;
    }
  }
  #pragma unroll
  for (int k = 0; k < KNN; k++) { cv[tid*KNN+k] = tv[k]; ci[tid*KNN+k] = ti[k]; }
  __syncthreads();

  for (int k = 0; k < KNN; k++) {
    float bv = 3.4e38f; int bi = 0x7fffffff; int bp = 0;
    #pragma unroll
    for (int i = 0; i < KNN; i++) {
      int e = i * 256 + tid;
      float v = cv[e];
      if (v < bv || (v == bv && ci[e] < bi)) { bv = v; bi = ci[e]; bp = e; }
    }
    rv[tid] = bv; ri[tid] = bi; rp[tid] = bp;
    __syncthreads();
    for (int s = 128; s > 0; s >>= 1) {
      if (tid < s) {
        if (rv[tid+s] < rv[tid] || (rv[tid+s] == rv[tid] && ri[tid+s] < ri[tid])) {
          rv[tid] = rv[tid+s]; ri[tid] = ri[tid+s]; rp[tid] = rp[tid+s];
        }
      }
      __syncthreads();
    }
    if (tid == 0) { knn_out[bm*KNN + k] = ri[0]; cv[rp[0]] = 3.4e38f; }
    __syncthreads();
  }
}

// ---------------------------------------------------------------------------
// Gather 16 neighbor feature rows (bf16), max-pool -> pooled bf16 [B*M, 768]
// ---------------------------------------------------------------------------
__global__ __launch_bounds__(256) void gather_maxpool(
    const unsigned short* __restrict__ pf, const int* __restrict__ knn,
    unsigned short* __restrict__ pooled)
{
  const int bm  = blockIdx.x;
  const int b   = bm >> 7;
  const int tid = threadIdx.x;
  __shared__ int nb[KNN];
  if (tid < KNN) nb[tid] = knn[bm*KNN + tid];
  __syncthreads();
  const size_t base = (size_t)b * NPTS;
  for (int c = 0; c < DTOK / 256; c++) {
    int d = c * 256 + tid;
    float v = -3.4e38f;
    #pragma unroll
    for (int k = 0; k < KNN; k++) {
      float x = bf16_to_f32(pf[(base + nb[k]) * (size_t)DTOK + d]);
      v = fmaxf(v, x);
    }
    pooled[(size_t)bm * DTOK + d] = f32_to_bf16(v);
  }
}

// ---------------------------------------------------------------------------
// Rank tokens by centroid time (stable), scatter tokens/centroids/masks.
// ---------------------------------------------------------------------------
__global__ __launch_bounds__(128) void sort_scatter(
    const float* __restrict__ cent, const float* __restrict__ tok,
    float* __restrict__ out_tok, float* __restrict__ out_cent,
    float* __restrict__ out_mask)
{
  const int b = blockIdx.x;
  const int i = threadIdx.x;
  __shared__ float st[MTOK];
  st[i] = cent[(b*MTOK + i) * 4 + 3];
  __syncthreads();
  float ti_ = st[i];
  int rank = 0;
  for (int j = 0; j < MTOK; j++) {
    float tj = st[j];
    if (tj < ti_ || (tj == ti_ && j < i)) rank++;
  }
  const size_t src = (size_t)(b*MTOK + i);
  const size_t dst = (size_t)(b*MTOK + rank);
  #pragma unroll
  for (int c = 0; c < 4; c++) out_cent[dst*4 + c] = cent[src*4 + c];
  out_mask[b*MTOK + i] = 1.0f;
  for (int d = 0; d < DTOK; d++) out_tok[dst*DTOK + d] = tok[src*DTOK + d];
}

// ---------------------------------------------------------------------------
extern "C" void kernel_launch(void* const* d_in, const int* in_sizes, int n_in,
                              void* d_out, int out_size, void* d_ws, size_t ws_size,
                              hipStream_t stream)
{
  const float* coords   = (const float*)d_in[0];
  const float* features = (const float*)d_in[1];
  const float* W1 = (const float*)d_in[2];   const float* b1 = (const float*)d_in[3];
  const float* W2 = (const float*)d_in[4];   const float* b2 = (const float*)d_in[5];
  const float* W3 = (const float*)d_in[6];   const float* b3 = (const float*)d_in[7];
  const float* W4 = (const float*)d_in[8];   const float* b4 = (const float*)d_in[9];
  const float* Wn1 = (const float*)d_in[10]; const float* bn1 = (const float*)d_in[11];
  const float* Wn2 = (const float*)d_in[12]; const float* bn2 = (const float*)d_in[13];
  (void)in_sizes; (void)n_in; (void)out_size; (void)ws_size;

  const int Bn = B_BATCH;
  const int MR = Bn * NPTS;                  // 131072 point rows

  char*  ws  = (char*)d_ws;
  size_t off = 0;
  auto alloc = [&](size_t bytes) {
    size_t o = off; off += (bytes + 255) & ~(size_t)255; return (void*)(ws + o);
  };
  unsigned short* actA   = (unsigned short*)alloc((size_t)MR * 768 * 2);
  unsigned short* actB   = (unsigned short*)alloc((size_t)MR * 768 * 2);
  unsigned short* pooled = (unsigned short*)alloc((size_t)Bn * MTOK * DTOK * 2);
  unsigned short* tmid   = (unsigned short*)alloc((size_t)Bn * MTOK * DTOK * 2);
  float*          tok    = (float*)alloc((size_t)Bn * MTOK * DTOK * 4);
  int*            fidx   = (int*)alloc((size_t)Bn * MTOK * 4);
  float*          cent   = (float*)alloc((size_t)Bn * MTOK * 4 * 4);
  int*            knn    = (int*)alloc((size_t)Bn * MTOK * KNN * 4);

  // per-point MLP: 32 -> 256 -> 512 -> 768 -> 768 (ReLU between, last plain)
  gemm_bias_act<float,          unsigned short, true ><<<dim3(256/64, MR/256), 256, 0, stream>>>(features, W1, b1, actA, MR, 256, 32);
  gemm_bias_act<unsigned short, unsigned short, true ><<<dim3(512/64, MR/256), 256, 0, stream>>>(actA, W2, b2, actB, MR, 512, 256);
  gemm_bias_act<unsigned short, unsigned short, true ><<<dim3(768/64, MR/256), 256, 0, stream>>>(actB, W3, b3, actA, MR, 768, 512);
  gemm_bias_act<unsigned short, unsigned short, false><<<dim3(768/64, MR/256), 256, 0, stream>>>(actA, W4, b4, actB, MR, 768, 768);

  fps_kernel<<<Bn, 1024, 0, stream>>>(coords, features, fidx, cent);
  knn_kernel<<<Bn * MTOK, 256, 0, stream>>>(coords, features, cent, knn);
  gather_maxpool<<<Bn * MTOK, 256, 0, stream>>>(actB, knn, pooled);

  // token MLP: 768 -> 768 (ReLU) -> 768
  gemm_bias_act<unsigned short, unsigned short, true ><<<dim3(768/64, (Bn*MTOK)/256), 256, 0, stream>>>(pooled, Wn1, bn1, tmid, Bn*MTOK, 768, 768);
  gemm_bias_act<unsigned short, float,          false><<<dim3(768/64, (Bn*MTOK)/256), 256, 0, stream>>>(tmid, Wn2, bn2, tok, Bn*MTOK, 768, 768);

  float* out_tok  = (float*)d_out;
  float* out_cent = out_tok  + (size_t)Bn * MTOK * DTOK;
  float* out_mask = out_cent + (size_t)Bn * MTOK * 4;
  sort_scatter<<<Bn, MTOK, 0, stream>>>(cent, tok, out_tok, out_cent, out_mask);
}
